// SelfAttention2D_24223615550132
// MI455X (gfx1250) — matrix operand order repaired and checked by reading, compiled, not run on hardware
//
#include <hip/hip_runtime.h>

typedef _Float16 h16;
typedef __attribute__((ext_vector_type(16))) _Float16 v16h;
typedef __attribute__((ext_vector_type(8)))  _Float16 v8h;
typedef __attribute__((ext_vector_type(8)))  float    v8f;

#define BATCH 8
#define CCH   256
#define SPT   1024      // H*W = 32*32
#define NH    8
#define HD    32
#define NGRP  32
#define CPG   8         // CCH / NGRP
#define LOG2E 1.44269504088896f

union V16H { v16h v; v8h h2[2]; h16 e[16]; int i[8]; };
union V8F  { v8f  v; float e[8]; };

// ---------------- WMMA fragment helpers (CDNA5 16x16x32 f16 layouts) ----------------

// A-fragment, 16x32 (MxK) f16, from row-major src (row stride in halves).
// lanes 0-15: M=lane, K=0..15; lanes 16-31: M=lane-16, K=16..31
__device__ __forceinline__ v16h load_a_frag(const h16* base, int row_stride,
                                            int row0, int col0, int lane) {
  int row  = row0 + (lane & 15);
  int kofs = (lane >> 4) << 4;          // 0 or 16
  const h16* p = base + (size_t)row * row_stride + col0 + kofs;
  V16H r;
  r.h2[0] = *(const v8h*)(p);           // K = col0+kofs .. +7
  r.h2[1] = *(const v8h*)(p + 8);       // K = col0+kofs+8 .. +15
  return r.v;
}

// B-fragment, 32x16 (KxN) f16, where B[k][n] = srcN[n][k] (srcN is N-major, row
// stride in halves). lanes 0-15: N=lane, K=0..15; lanes 16-31: N=lane-16, K=16..31
__device__ __forceinline__ v16h load_b_frag(const h16* srcN, int row_stride,
                                            int n0, int k0, int lane) {
  const h16* p = srcN + (size_t)(n0 + (lane & 15)) * row_stride + k0 + ((lane >> 4) << 4);
  V16H r;
  r.h2[0] = *(const v8h*)(p);
  r.h2[1] = *(const v8h*)(p + 8);
  return r.v;
}

__device__ __forceinline__ v8f wmma16(v16h a, v16h b, v8f c) {
  return __builtin_amdgcn_wmma_f32_16x16x32_f16(false, a, false, b, (short)0, c, false, false);
}

// ---------------- Kernel 0: f32 -> f16 weight convert ----------------

__global__ void cvt_f32_f16_kernel(const float* __restrict__ in, h16* __restrict__ out, int n) {
  int i = blockIdx.x * 256 + threadIdx.x;
  if (i < n) out[i] = (h16)in[i];
}

// ---------------- Kernel 1: GroupNorm, writes token-major f16 xn[b][s][c] ----------------

__global__ __launch_bounds__(256) void gnorm_kernel(const float* __restrict__ x,
                                                    const float* __restrict__ gw,
                                                    const float* __restrict__ gb,
                                                    h16* __restrict__ xnh) {
  int b = blockIdx.x / NGRP;
  int g = blockIdx.x % NGRP;
  int tid = threadIdx.x;
  const float* xg = x + ((size_t)b * CCH + g * CPG) * SPT;   // 8 ch * 1024 px
  float vals[32];
  float sum = 0.f, sq = 0.f;
#pragma unroll
  for (int i = 0; i < 32; ++i) {
    float v = xg[tid + i * 256];
    vals[i] = v; sum += v; sq += v * v;
  }
#pragma unroll
  for (int off = 16; off > 0; off >>= 1) {
    sum += __shfl_xor(sum, off, 32);
    sq  += __shfl_xor(sq,  off, 32);
  }
  __shared__ float ls[8], lq[8];
  int wid = tid >> 5;
  if ((tid & 31) == 0) { ls[wid] = sum; lq[wid] = sq; }
  __syncthreads();
  sum = 0.f; sq = 0.f;
#pragma unroll
  for (int i = 0; i < 8; ++i) { sum += ls[i]; sq += lq[i]; }
  float mu  = sum * (1.f / 8192.f);
  float var = sq * (1.f / 8192.f) - mu * mu;
  float rs  = rsqrtf(var + 1e-5f);
#pragma unroll
  for (int i = 0; i < 32; ++i) {
    int idx = tid + i * 256;
    int cl  = idx >> 10;         // local channel 0..7
    int s   = idx & 1023;
    int c   = g * CPG + cl;
    float xn = (vals[i] - mu) * rs * gw[c] + gb[c];
    xnh[((size_t)b * SPT + s) * CCH + c] = (h16)xn;
  }
}

// ---------------- Kernel 2: QKV GEMM  D[o][s] = qkv_w . xn^T ----------------
// One wave per block; 16(o) x 64(s) tile; K=256 -> 32 WMMAs.
// Epilogue scatters: q[b][h][s][d] (prescaled 1/sqrt(hd)), k[b][h][t][d], v^T[b][h][d][t]

__global__ __launch_bounds__(32) void qkv_gemm_kernel(const h16* __restrict__ xnh,
                                                      const h16* __restrict__ wh,
                                                      const float* __restrict__ qkv_b,
                                                      h16* __restrict__ qh,
                                                      h16* __restrict__ kh,
                                                      h16* __restrict__ vth) {
  int lane = threadIdx.x;
  int s0 = blockIdx.x * 64;
  int o0 = blockIdx.y * 16;
  int b  = blockIdx.z;
  const h16* bbase = xnh + (size_t)b * SPT * CCH;
  v8f acc0 = {}, acc1 = {}, acc2 = {}, acc3 = {};
#pragma unroll
  for (int kk = 0; kk < CCH; kk += 32) {
    v16h aw = load_a_frag(wh, CCH, o0, kk, lane);
    acc0 = wmma16(aw, load_b_frag(bbase, CCH, s0 +  0, kk, lane), acc0);
    acc1 = wmma16(aw, load_b_frag(bbase, CCH, s0 + 16, kk, lane), acc1);
    acc2 = wmma16(aw, load_b_frag(bbase, CCH, s0 + 32, kk, lane), acc2);
    acc3 = wmma16(aw, load_b_frag(bbase, CCH, s0 + 48, kk, lane), acc3);
  }
  int half  = lane >> 4;
  int sl    = lane & 15;
  int which = o0 >> 8;               // 0=q 1=k 2=v (uniform per block)
  int oc0   = o0 & 255;
  int hh    = oc0 >> 5;              // head (uniform)
  int dbase = (oc0 & 31) + 8 * half; // d for r=0
  v8f accs[4] = {acc0, acc1, acc2, acc3};
#pragma unroll
  for (int j = 0; j < 4; ++j) {
    V8F a; a.v = accs[j];
    int s = s0 + 16 * j + sl;
#pragma unroll
    for (int r = 0; r < 8; ++r) {
      int o = o0 + r + 8 * half;
      float w = a.e[r] + qkv_b[o];
      int d = dbase + r;
      if (which == 0)
        qh[(((size_t)b * NH + hh) * SPT + s) * HD + d] = (h16)(w * 0.1767766953f);
      else if (which == 1)
        kh[(((size_t)b * NH + hh) * SPT + s) * HD + d] = (h16)w;
      else
        vth[(((size_t)b * NH + hh) * HD + d) * SPT + s] = (h16)w;
    }
  }
}

// ---------------- Kernel 3: flash attention, one wave = 16 queries ----------------
// S^T = K.Q^T (2 WMMA per 32-key tile), online softmax (per-lane row stats since
// lane owns query s = lane&15), P^T B-fragment built with 8 xor-16 shuffles,
// out^T[d][s] += V^T.P^T (2 WMMA).  Writes att[b][s][h*32+d] (f16, row-major).

__global__ __launch_bounds__(32) void attn_kernel(const h16* __restrict__ qh,
                                                  const h16* __restrict__ kh,
                                                  const h16* __restrict__ vth,
                                                  h16* __restrict__ att) {
  int lane = threadIdx.x;
  int s0 = blockIdx.x * 16;
  int h  = blockIdx.y;
  int b  = blockIdx.z;
  const h16* qbase = qh  + (size_t)(b * NH + h) * SPT * HD;
  const h16* kbase = kh  + (size_t)(b * NH + h) * SPT * HD;
  const h16* vbase = vth + (size_t)(b * NH + h) * HD * SPT;

  v16h bq = load_b_frag(qbase, HD, s0, 0, lane);   // B = Q^T (d x s)
  v8f accT0 = {}, accT1 = {};                      // out^T: d=0..15 / d=16..31, N=s
  float m = -__builtin_inff();
  float lsum = 0.f;
  v8f zero = {};
  bool hiHalf = lane >= 16;

#pragma unroll 1
  for (int t0 = 0; t0 < SPT; t0 += 32) {
    v16h ak0 = load_a_frag(kbase, HD, t0,      0, lane);
    v16h ak1 = load_a_frag(kbase, HD, t0 + 16, 0, lane);
    V8F st0; st0.v = wmma16(ak0, bq, zero);        // S^T[t0+  r+8*half][s=lane&15]
    V8F st1; st1.v = wmma16(ak1, bq, zero);        // S^T[t0+16+r+8*half][s]

    // row max over the 32 keys of this tile (16 in-lane + 16 via xor-16)
    float mx = st0.e[0];
#pragma unroll
    for (int i = 1; i < 8; ++i) mx = fmaxf(mx, st0.e[i]);
#pragma unroll
    for (int i = 0; i < 8; ++i) mx = fmaxf(mx, st1.e[i]);
    mx = fmaxf(mx, __shfl_xor(mx, 16, 32));
    float mnew  = fmaxf(m, mx);
    float alpha = exp2f((m - mnew) * LOG2E);

    union { int i[4]; h16 e[8]; } pk0, pk1;
    float psum = 0.f;
#pragma unroll
    for (int i = 0; i < 8; ++i) {
      float e0 = exp2f((st0.e[i] - mnew) * LOG2E);
      float e1 = exp2f((st1.e[i] - mnew) * LOG2E);
      pk0.e[i] = (h16)e0; pk1.e[i] = (h16)e1;
      psum += e0 + e1;
    }
    psum += __shfl_xor(psum, 16, 32);
    lsum = lsum * alpha + psum;
    m = mnew;
#pragma unroll
    for (int i = 0; i < 8; ++i) { accT0[i] *= alpha; accT1[i] *= alpha; }

    // Build B-fragment P^T (K=t 32 x N=s 16):
    //  lane<16 needs t0+0..15 for s=lane   -> {own pk0, xor16(pk0)}
    //  lane>=16 needs t0+16..31 for s=l-16 -> {xor16(pk1), own pk1}
    V16H bp;
#pragma unroll
    for (int i = 0; i < 4; ++i) {
      int o0i = __shfl_xor(pk0.i[i], 16, 32);
      int o1i = __shfl_xor(pk1.i[i], 16, 32);
      bp.i[i]     = hiHalf ? o1i : pk0.i[i];
      bp.i[4 + i] = hiHalf ? pk1.i[i] : o0i;
    }

    v16h av0 = load_a_frag(vbase, SPT, 0,  t0, lane);  // V^T rows d=0..15, K=t
    v16h av1 = load_a_frag(vbase, SPT, 16, t0, lane);  // V^T rows d=16..31
    accT0 = wmma16(av0, bp.v, accT0);
    accT1 = wmma16(av1, bp.v, accT1);
  }

  float inv_l = 1.f / lsum;                 // per-lane uniform (s = lane&15)
  int s = s0 + (lane & 15);
  int dofs = 8 * (lane >> 4);
  h16* abase = att + ((size_t)b * SPT + s) * CCH + h * HD;
  V8F a0; a0.v = accT0;
  V8F a1; a1.v = accT1;
#pragma unroll
  for (int r = 0; r < 8; ++r) {
    abase[r + dofs]      = (h16)(a0.e[r] * inv_l);
    abase[16 + r + dofs] = (h16)(a1.e[r] * inv_l);
  }
}

// ---------------- Kernel 4: out projection + bias + residual ----------------
// D[o][s] = out_w . att^T ; y[b][o][s] = x[b][o][s] + D + out_b[o]

__global__ __launch_bounds__(32) void proj_kernel(const h16* __restrict__ att,
                                                  const h16* __restrict__ wh,
                                                  const float* __restrict__ out_b,
                                                  const float* __restrict__ x,
                                                  float* __restrict__ y) {
  int lane = threadIdx.x;
  int s0 = blockIdx.x * 64;
  int o0 = blockIdx.y * 16;
  int b  = blockIdx.z;
  const h16* bbase = att + (size_t)b * SPT * CCH;
  v8f acc0 = {}, acc1 = {}, acc2 = {}, acc3 = {};
#pragma unroll
  for (int kk = 0; kk < CCH; kk += 32) {
    v16h aw = load_a_frag(wh, CCH, o0, kk, lane);
    acc0 = wmma16(aw, load_b_frag(bbase, CCH, s0 +  0, kk, lane), acc0);
    acc1 = wmma16(aw, load_b_frag(bbase, CCH, s0 + 16, kk, lane), acc1);
    acc2 = wmma16(aw, load_b_frag(bbase, CCH, s0 + 32, kk, lane), acc2);
    acc3 = wmma16(aw, load_b_frag(bbase, CCH, s0 + 48, kk, lane), acc3);
  }
  int half = lane >> 4, sl = lane & 15;
  v8f accs[4] = {acc0, acc1, acc2, acc3};
#pragma unroll
  for (int j = 0; j < 4; ++j) {
    V8F a; a.v = accs[j];
    int s = s0 + 16 * j + sl;
#pragma unroll
    for (int r = 0; r < 8; ++r) {
      int o = o0 + r + 8 * half;
      size_t idx = ((size_t)b * CCH + o) * SPT + s;
      y[idx] = x[idx] + a.e[r] + out_b[o];
    }
  }
}

// ---------------- Launch ----------------

extern "C" void kernel_launch(void* const* d_in, const int* in_sizes, int n_in,
                              void* d_out, int out_size, void* d_ws, size_t ws_size,
                              hipStream_t stream) {
  const float* x     = (const float*)d_in[0];
  const float* gn_w  = (const float*)d_in[1];
  const float* gn_b  = (const float*)d_in[2];
  const float* qkv_w = (const float*)d_in[3];
  const float* qkv_b = (const float*)d_in[4];
  const float* out_w = (const float*)d_in[5];
  const float* out_b = (const float*)d_in[6];
  float* y = (float*)d_out;

  char* ws = (char*)d_ws;
  h16* xnh   = (h16*)(ws);                                   // 4 MiB  xn[b][s][c]
  h16* qh    = (h16*)(ws + ( 4u << 20));                     // 4 MiB  q[b][h][s][d]
  h16* kh    = (h16*)(ws + ( 8u << 20));                     // 4 MiB  k[b][h][t][d]
  h16* vth   = (h16*)(ws + (12u << 20));                     // 4 MiB  v^T[b][h][d][t]
  h16* att   = (h16*)(ws + (16u << 20));                     // 4 MiB  att[b][s][c]
  h16* qkvwh = (h16*)(ws + (20u << 20));                     // 384 KiB
  h16* outwh = (h16*)(ws + (20u << 20) + (1u << 19));        // 128 KiB

  cvt_f32_f16_kernel<<<dim3(768), 256, 0, stream>>>(qkv_w, qkvwh, 3 * CCH * CCH);
  cvt_f32_f16_kernel<<<dim3(256), 256, 0, stream>>>(out_w, outwh, CCH * CCH);

  gnorm_kernel<<<dim3(BATCH * NGRP), 256, 0, stream>>>(x, gn_w, gn_b, xnh);

  qkv_gemm_kernel<<<dim3(SPT / 64, (3 * CCH) / 16, BATCH), 32, 0, stream>>>(
      xnh, qkvwh, qkv_b, qh, kh, vth);

  attn_kernel<<<dim3(SPT / 16, NH, BATCH), 32, 0, stream>>>(qh, kh, vth, att);

  proj_kernel<<<dim3(SPT / 64, CCH / 16, BATCH), 32, 0, stream>>>(
      att, outwh, out_b, x, y);
}
